// MT_Attention_58523224375748
// MI455X (gfx1250) — compile-verified, matrix operand order checked by reading, not run on hardware
//
#include <hip/hip_runtime.h>

#define HIDC 4096
#define NH   32
#define HD   128

typedef __attribute__((ext_vector_type(16))) __bf16 v16bf;
typedef __attribute__((ext_vector_type(8)))  float  v8f;

// ---------------- segment ids from int64 seqlen arrays ----------------
__global__ void seg_init_kernel(const long long* __restrict__ qlen,
                                const long long* __restrict__ klen,
                                int* __restrict__ segq, int* __restrict__ segk,
                                int S, int Sk)
{
    __shared__ int qo[4], ko[4];
    if (threadIdx.x == 0) {
        int a = 0, b = 0;
        for (int i = 0; i < 4; ++i) { a += (int)qlen[i]; qo[i] = a; b += (int)klen[i]; ko[i] = b; }
    }
    __syncthreads();
    for (int s = threadIdx.x; s < S; s += blockDim.x) {
        int g = 0;
        for (int i = 0; i < 3; ++i) if (s >= qo[i]) g = i + 1;
        segq[s] = g;
    }
    for (int s = threadIdx.x; s < Sk; s += blockDim.x) {
        int g = 0;
        for (int i = 0; i < 3; ++i) if (s >= ko[i]) g = i + 1;
        segk[s] = g;
    }
}

// ---------------- fp32 -> bf16 straight convert ----------------
__global__ __launch_bounds__(256)
void cvt_f32_bf16(const float* __restrict__ src, __bf16* __restrict__ dst, long long n)
{
    long long i = ((long long)blockIdx.x * 256 + threadIdx.x) * 4;
    if (i + 3 < n) {
        float4 v = *(const float4*)(src + i);
        dst[i + 0] = (__bf16)v.x; dst[i + 1] = (__bf16)v.y;
        dst[i + 2] = (__bf16)v.z; dst[i + 3] = (__bf16)v.w;
    }
}

// ---------------- fp32 (rows x cols) -> bf16 transposed (cols x rows) ----------------
__global__ __launch_bounds__(256)
void transpose_cvt_kernel(const float* __restrict__ src, __bf16* __restrict__ dst,
                          int rows, int cols)
{
    __shared__ float tile[32][33];
    const int c0 = blockIdx.x * 32;
    const int r0 = blockIdx.y * 32;
    for (int j = threadIdx.y; j < 32; j += 8)
        tile[j][threadIdx.x] = src[(long long)(r0 + j) * cols + (c0 + threadIdx.x)];
    __syncthreads();
    for (int j = threadIdx.y; j < 32; j += 8)
        dst[(long long)(c0 + j) * rows + (r0 + threadIdx.x)] = (__bf16)tile[threadIdx.x][j];
}

// ---------------- generic bf16 WMMA GEMM:  C(MxN) = A(MxK) * BT(NxK)^T ----------------
// Block = 128 threads (4 waves, 2x2), wave tile = 64x64 -> block tile 128x128.
// 16 accumulators x 16 WMMA per 32-K step, 8 operand vectors -> 32 FLOP/byte from L2.
// mode 0: fp32 row-major to out0   (ldc = N)
// mode 1: bf16 row-major to out0   (ldc = N)
// mode 2: col n < split  -> bf16 head-major [h][s][d] to out0, scaled
//         col n >= split -> bf16 transposed [h][d][s] to out1 (for V^T)
// All of M, N divisible by 128 and K divisible by 32 in this problem.
__global__ __launch_bounds__(128)
void gemm_wmma_bf16(const __bf16* __restrict__ A, const __bf16* __restrict__ BT,
                    void* __restrict__ out0, void* __restrict__ out1,
                    int N, int K, int mode, float scale, int split, int Scols,
                    long long sA, long long sB, long long sC)
{
    const int z = blockIdx.z;
    A  += (long long)z * sA;
    BT += (long long)z * sB;

    const int wave = threadIdx.x >> 5;
    const int lane = threadIdx.x & 31;
    const int m0 = blockIdx.y * 128 + (wave & 1) * 64;
    const int n0 = blockIdx.x * 128 + (wave >> 1) * 64;
    const int r    = lane & 15;          // row within 16-tile
    const int koff = (lane >> 4) << 4;   // lanes 16-31 take K+16..K+31

    const __bf16* aP[4];
    const __bf16* bP[4];
    #pragma unroll
    for (int i = 0; i < 4; ++i) {
        aP[i] = A  + (long long)(m0 + 16 * i + r) * K + koff;
        bP[i] = BT + (long long)(n0 + 16 * i + r) * K + koff;
    }

    v8f acc[4][4] = {};

    for (int k0 = 0; k0 < K; k0 += 32) {
        if ((k0 & 127) == 0) {          // one prefetch per cacheline span per stream
            #pragma unroll
            for (int i = 0; i < 4; ++i) {
                __builtin_prefetch(aP[i] + k0 + 512, 0, 3);
                __builtin_prefetch(bP[i] + k0 + 512, 0, 3);
            }
        }
        v16bf av[4], bv[4];
        #pragma unroll
        for (int i = 0; i < 4; ++i) {
            av[i] = *(const v16bf*)(aP[i] + k0);
            bv[i] = *(const v16bf*)(bP[i] + k0);
        }
        #pragma unroll
        for (int i = 0; i < 4; ++i)
            #pragma unroll
            for (int j = 0; j < 4; ++j)
                acc[i][j] = __builtin_amdgcn_wmma_f32_16x16x32_bf16(
                    false, av[i], false, bv[j], (short)0, acc[i][j], false, false);
    }

    // C/D layout: VGPR e, lanes 0-15 -> (M=e, N=lane); lanes 16-31 -> (M=e+8, N=lane-16)
    const int mloc = (lane >> 4) * 8;
    const int nloc = lane & 15;

    #pragma unroll
    for (int i = 0; i < 4; ++i) {
        #pragma unroll
        for (int j = 0; j < 4; ++j) {
            const v8f& acv = acc[i][j];
            const int mb = m0 + 16 * i + mloc;
            const int nn = n0 + 16 * j + nloc;
            if (mode == 0) {
                float* C = (float*)out0 + (long long)z * sC;
                #pragma unroll
                for (int e = 0; e < 8; ++e) C[(long long)(mb + e) * N + nn] = acv[e];
            } else if (mode == 1) {
                __bf16* C = (__bf16*)out0 + (long long)z * sC;
                #pragma unroll
                for (int e = 0; e < 8; ++e) C[(long long)(mb + e) * N + nn] = (__bf16)acv[e];
            } else {
                if (nn < split) {
                    __bf16* O = (__bf16*)out0;
                    const int h = nn >> 7, d = nn & 127;
                    #pragma unroll
                    for (int e = 0; e < 8; ++e)
                        O[((long long)h * Scols + (mb + e)) * HD + d] = (__bf16)(acv[e] * scale);
                } else {
                    __bf16* O = (__bf16*)out1;
                    const int nv = nn - split;
                    const int h = nv >> 7, d = nv & 127;
                    #pragma unroll
                    for (int e = 0; e < 8; ++e)
                        O[((long long)h * HD + d) * Scols + (mb + e)] = (__bf16)acv[e];
                }
            }
        }
    }
}

// ---------------- grouped 4x9 conv + mask bias + softmax (one block per (q,h)) --------
__global__ __launch_bounds__(256)
void conv_mask_softmax(const __bf16* __restrict__ attn, const float* __restrict__ convw,
                       const int* __restrict__ segq, const int* __restrict__ segk,
                       __bf16* __restrict__ probs, int S, int Sk)
{
    const int q = blockIdx.x;
    const int h = blockIdx.y;
    const int g = h >> 1;
    __shared__ float w[72];               // [c][dq][dk] = c*36 + dq*9 + dk
    __shared__ float rows[8][1544];       // [c*4+dq][k+4], S<=1536 with +-4 col halo
    __shared__ float red[8];
    const int tid  = threadIdx.x;
    const int lane = tid & 31;
    const int wid  = tid >> 5;

    if (tid < 72) w[tid] = convw[h * 72 + tid];
    const int sgq = segq[q];

    // Stage 2 channels x 4 rows (q-2..q+1) with zero pad and block-diagonal mask
    for (int idx = tid; idx < 8 * 1544; idx += 256) {
        const int rr = idx / 1544;
        const int kk = idx - rr * 1544;
        const int c  = rr >> 2;
        const int j  = rr & 3;
        const int qi = q + j - 2;
        const int ki = kk - 4;
        float v = 0.f;
        if (qi >= 0 && qi < S && ki >= 0 && ki < Sk && segq[qi] == segk[ki])
            v = (float)attn[((long long)(g * 2 + c) * S + qi) * Sk + ki];
        rows[rr][kk] = v;
    }
    __syncthreads();

    float vals[6];
    float lmax = -3.0e38f;
    int nt = 0;
    for (int k = tid; k < Sk; k += 256, ++nt) {
        float acc = 0.f;
        #pragma unroll
        for (int c = 0; c < 2; ++c)
            #pragma unroll
            for (int j = 0; j < 4; ++j) {
                const float* rp = &rows[c * 4 + j][k];   // input col k-4+dk -> [k+dk]
                const float* wp = &w[(c * 4 + j) * 9];
                #pragma unroll
                for (int dk = 0; dk < 9; ++dk) acc = fmaf(wp[dk], rp[dk], acc);
            }
        if (sgq == segk[k]) acc += 1.0f;                 // softmax(conv + mask)
        vals[nt] = acc;
        lmax = fmaxf(lmax, acc);
    }
    #pragma unroll
    for (int off = 16; off > 0; off >>= 1) lmax = fmaxf(lmax, __shfl_xor(lmax, off));
    if (lane == 0) red[wid] = lmax;
    __syncthreads();
    float bmax = red[0];
    #pragma unroll
    for (int i = 1; i < 8; ++i) bmax = fmaxf(bmax, red[i]);
    __syncthreads();

    float lsum = 0.f;
    for (int t = 0; t < nt; ++t) { vals[t] = __expf(vals[t] - bmax); lsum += vals[t]; }
    #pragma unroll
    for (int off = 16; off > 0; off >>= 1) lsum += __shfl_xor(lsum, off);
    if (lane == 0) red[wid] = lsum;
    __syncthreads();
    float bsum = 0.f;
    #pragma unroll
    for (int i = 0; i < 8; ++i) bsum += red[i];
    const float inv = 1.0f / bsum;

    nt = 0;
    for (int k = tid; k < Sk; k += 256, ++nt)
        probs[((long long)h * S + q) * Sk + k] = (__bf16)(vals[nt] * inv);
}

// ---------------- group-norm stats: one block per group of 2 heads ----------------
__global__ __launch_bounds__(256)
void gn_stats_kernel(const float* __restrict__ o, float* __restrict__ mu,
                     float* __restrict__ rinv, int S)
{
    const int g = blockIdx.x;
    const float* base = o + (long long)g * 2 * S * HD;
    const long long n = (long long)2 * S * HD;
    float s = 0.f, s2 = 0.f;
    for (long long i = threadIdx.x; i < n; i += 256) {
        float v = base[i];
        s += v; s2 = fmaf(v, v, s2);
    }
    #pragma unroll
    for (int off = 16; off > 0; off >>= 1) { s += __shfl_xor(s, off); s2 += __shfl_xor(s2, off); }
    __shared__ float rs[8], rs2[8];
    const int lane = threadIdx.x & 31, wid = threadIdx.x >> 5;
    if (lane == 0) { rs[wid] = s; rs2[wid] = s2; }
    __syncthreads();
    if (threadIdx.x == 0) {
        float a = 0.f, b = 0.f;
        for (int i = 0; i < 8; ++i) { a += rs[i]; b += rs2[i]; }
        const float m = a / (float)n;
        mu[g]   = m;
        rinv[g] = rsqrtf(b / (float)n - m * m + 1e-5f);
    }
}

// ---------------- apply gn + transpose to (S, H*D) bf16 for final GEMM ----------------
__global__ __launch_bounds__(256)
void gn_apply_kernel(const float* __restrict__ o, const float* __restrict__ mu,
                     const float* __restrict__ rinv,
                     const float* __restrict__ gw, const float* __restrict__ gb,
                     __bf16* __restrict__ out2, int S)
{
    const long long i = (long long)blockIdx.x * 256 + threadIdx.x;
    const int d = (int)(i & (HD - 1));
    const long long t = i >> 7;
    const int s = (int)(t % S);
    const int h = (int)(t / S);
    const int g = h >> 1;
    const float v = (o[i] - mu[g]) * rinv[g] * gw[h] + gb[h];
    out2[(long long)s * HIDC + h * HD + d] = (__bf16)v;
}

extern "C" void kernel_launch(void* const* d_in, const int* in_sizes, int n_in,
                              void* d_out, int out_size, void* d_ws, size_t ws_size,
                              hipStream_t stream)
{
    const float* x   = (const float*)d_in[0];
    const float* ctx = (const float*)d_in[1];
    const float* wq  = (const float*)d_in[2];
    const float* wkv = (const float*)d_in[3];
    const float* wo  = (const float*)d_in[4];
    const float* cw  = (const float*)d_in[5];
    const float* gnw = (const float*)d_in[6];
    const float* gnb = (const float*)d_in[7];
    const long long* qlen = (const long long*)d_in[8];
    const long long* klen = (const long long*)d_in[9];

    const int S  = in_sizes[0] / HIDC;   // 1536
    const int Sk = in_sizes[1] / HIDC;   // 1536
    (void)n_in; (void)out_size; (void)ws_size;

    // workspace carve-out (256B aligned); total ~540 MB
    char* base = (char*)d_ws;
    size_t off = 0;
    auto wsalloc = [&](size_t bytes) -> char* {
        char* p = base + off;
        off = (off + bytes + 255) & ~(size_t)255;
        return p;
    };
    __bf16* xb    = (__bf16*)wsalloc((size_t)S  * HIDC * 2);
    __bf16* cb    = (__bf16*)wsalloc((size_t)Sk * HIDC * 2);
    __bf16* wqT   = (__bf16*)wsalloc((size_t)HIDC * HIDC * 2);
    __bf16* wkvT  = (__bf16*)wsalloc((size_t)2 * HIDC * HIDC * 2);
    __bf16* woT   = (__bf16*)wsalloc((size_t)HIDC * HIDC * 2);
    __bf16* qb    = (__bf16*)wsalloc((size_t)NH * S  * HD * 2);
    __bf16* kb    = (__bf16*)wsalloc((size_t)NH * Sk * HD * 2);
    __bf16* vT    = (__bf16*)wsalloc((size_t)NH * HD * Sk * 2);
    __bf16* attn  = (__bf16*)wsalloc((size_t)NH * S * Sk * 2);
    __bf16* probs = (__bf16*)wsalloc((size_t)NH * S * Sk * 2);
    float*  pv    = (float*) wsalloc((size_t)NH * S * HD * 4);
    __bf16* out2  = (__bf16*)wsalloc((size_t)S * HIDC * 2);
    int*    segq  = (int*)   wsalloc((size_t)S * 4);
    int*    segk  = (int*)   wsalloc((size_t)Sk * 4);
    float*  mu    = (float*) wsalloc(16 * 4);
    float*  ri    = (float*) wsalloc(16 * 4);

    seg_init_kernel<<<1, 256, 0, stream>>>(qlen, klen, segq, segk, S, Sk);

    const long long nx = (long long)S * HIDC;
    cvt_f32_bf16<<<(int)(nx / 4 / 256), 256, 0, stream>>>(x, xb, nx);
    const long long nc = (long long)Sk * HIDC;
    cvt_f32_bf16<<<(int)(nc / 4 / 256), 256, 0, stream>>>(ctx, cb, nc);

    transpose_cvt_kernel<<<dim3(HIDC / 32, HIDC / 32), dim3(32, 8), 0, stream>>>(wq,  wqT,  HIDC, HIDC);
    transpose_cvt_kernel<<<dim3(2 * HIDC / 32, HIDC / 32), dim3(32, 8), 0, stream>>>(wkv, wkvT, HIDC, 2 * HIDC);
    transpose_cvt_kernel<<<dim3(HIDC / 32, HIDC / 32), dim3(32, 8), 0, stream>>>(wo,  woT,  HIDC, HIDC);

    const float scale = 0.08838834764831845f;   // 1/sqrt(128)
    // q = (x @ wq) * scale  -> head-major bf16
    gemm_wmma_bf16<<<dim3(HIDC / 128, S / 128, 1), 128, 0, stream>>>(
        xb, wqT, qb, nullptr, HIDC, HIDC, 2, scale, HIDC, S, 0, 0, 0);
    // kv = ctx @ wkv -> k head-major, v transposed [h][d][s]
    gemm_wmma_bf16<<<dim3(2 * HIDC / 128, Sk / 128, 1), 128, 0, stream>>>(
        cb, wkvT, kb, vT, 2 * HIDC, HIDC, 2, 1.0f, HIDC, Sk, 0, 0, 0);
    // attn[h] = q_h @ k_h^T  (bf16 row-major, batched over heads)
    gemm_wmma_bf16<<<dim3(Sk / 128, S / 128, NH), 128, 0, stream>>>(
        qb, kb, attn, nullptr, Sk, HD, 1, 1.0f, 0, 0,
        (long long)S * HD, (long long)Sk * HD, (long long)S * Sk);
    // grouped conv + mask + softmax -> probs bf16
    conv_mask_softmax<<<dim3(S, NH), 256, 0, stream>>>(attn, cw, segq, segk, probs, S, Sk);
    // out[h] = probs_h @ v_h  (fp32 head-major)
    gemm_wmma_bf16<<<dim3(HD / 128, S / 128, NH), 128, 0, stream>>>(
        probs, vT, pv, nullptr, HD, Sk, 0, 1.0f, 0, 0,
        (long long)S * Sk, (long long)HD * Sk, (long long)S * HD);
    // group norm
    gn_stats_kernel<<<16, 256, 0, stream>>>(pv, mu, ri, S);
    gn_apply_kernel<<<(NH * S * HD) / 256, 256, 0, stream>>>(pv, mu, ri, gnw, gnb, out2, S);
    // final projection -> d_out fp32 (S x HID)
    gemm_wmma_bf16<<<dim3(HIDC / 128, S / 128, 1), 128, 0, stream>>>(
        out2, woT, (float*)d_out, nullptr, HIDC, HIDC, 0, 1.0f, 0, 0, 0, 0, 0);
}
